// HModel_40604620816732
// MI455X (gfx1250) — matrix operand order repaired and checked by reading, not compiled, not run on hardware
//
#include <hip/hip_runtime.h>
#include <hip/hip_bf16.h>
#include <math.h>

typedef __attribute__((ext_vector_type(16))) _Float16 v16h;
typedef __attribute__((ext_vector_type(8)))  float    v8f;

#define BS   2
#define CH   256
#define HH   64
#define WW   64
#define SRR  8
#define MOO  17
#define NPL  289      // 17*17
#define HPAD 80       // H + 2*SR (padded rows of x2)
#define C1   144
#define C2   49
#define EPS_BN 1e-5f

// ---------------------------------------------------------------------------
// k1: per-pixel inverse L2 norm of f1 over channels (one wave per pixel)
// ---------------------------------------------------------------------------
__global__ __launch_bounds__(256) void k_norm(const float* __restrict__ f1,
                                              float* __restrict__ inv_norm) {
  int tid  = blockIdx.x * 256 + threadIdx.x;
  int wid  = tid >> 5, lane = tid & 31;          // wid in [0, BS*H*W)
  int w = wid % WW, h = (wid / WW) % HH, b = wid / (WW * HH);
  float s = 0.f;
  #pragma unroll
  for (int i = 0; i < CH / 32; ++i) {
    int c = lane + i * 32;
    float v = f1[(((size_t)b * CH + c) * HH + h) * WW + w];
    s += v * v;
  }
  #pragma unroll
  for (int off = 16; off >= 1; off >>= 1) s += __shfl_xor(s, off, 32);
  if (lane == 0) inv_norm[wid] = 1.f / fmaxf(sqrtf(s), 1e-12f);
}

// ---------------------------------------------------------------------------
// k2: pack normalized f1 into WMMA A-fragments (f16).
// A 16x32 layout: lane l<16 -> M=l, halves K..K+15;
//                 lane l>=16 -> M=l-16, halves K+16..K+31
// Apack[b][h][wt][kc][lane][16]
// ---------------------------------------------------------------------------
__global__ __launch_bounds__(256) void k_packA(const float* __restrict__ f1,
                                               const float* __restrict__ inv_norm,
                                               _Float16* __restrict__ Apack) {
  int tid = blockIdx.x * 256 + threadIdx.x;
  int lane = tid & 31;
  int kc = (tid >> 5) & 7;
  int wt = (tid >> 8) & 3;
  int h  = (tid >> 10) & 63;
  int b  = tid >> 16;
  int m  = lane & 15;
  int w  = wt * 16 + m;
  float inv = inv_norm[(b * HH + h) * WW + w];
  int khi = (lane >= 16) ? 16 : 0;
  _Float16* dst = Apack + ((((size_t)(b * HH + h) * 4 + wt) * 8 + kc) * 512) + lane * 16;
  #pragma unroll
  for (int j = 0; j < 16; ++j) {
    int c = kc * 32 + j + khi;
    dst[j] = (_Float16)(f1[(((size_t)b * CH + c) * HH + h) * WW + w] * inv);
  }
}

// ---------------------------------------------------------------------------
// k3: pack zero-padded f2 into WMMA B-fragments (f16).
// B 32x16 layout: lane l -> N=l%16; halves = K + (l<16 ? 0..15 : 16..31)
// Bpack[b][hp(80)][ct(5)][kc(8)][lane][16], column tile ct covers wp = ct*16..+15
// ---------------------------------------------------------------------------
__global__ __launch_bounds__(256) void k_packB(const float* __restrict__ f2,
                                               _Float16* __restrict__ Bpack) {
  int tid = blockIdx.x * 256 + threadIdx.x;
  int lane = tid & 31;
  int kc = (tid >> 5) & 7;
  int ct = (tid >> 8) % 5;
  int hp = (tid / 1280) % HPAD;
  int b  = tid / 102400;
  int n  = lane & 15;
  int wp = ct * 16 + n;
  int h2 = hp - SRR, w2 = wp - SRR;
  bool in = ((unsigned)h2 < HH) && ((unsigned)w2 < WW);
  int kof = (lane >= 16) ? 16 : 0;
  _Float16* dst = Bpack + ((((size_t)(b * HPAD + hp) * 5 + ct) * 8 + kc) * 512) + lane * 16;
  #pragma unroll
  for (int j = 0; j < 16; ++j) {
    int c = kc * 32 + kof + j;
    dst[j] = in ? (_Float16)f2[(((size_t)b * CH + c) * HH + h2) * WW + w2] : (_Float16)0.f;
  }
}

// ---------------------------------------------------------------------------
// k4: cost volume via WMMA. One wave per (b, dj, h, wtile): 16x32x256 GEMM,
// keep the di = wp - w in [0,16] band, /256 mean, leaky-relu, scatter to mv.
// ---------------------------------------------------------------------------
__global__ __launch_bounds__(256) void k_cv(const _Float16* __restrict__ Apack,
                                            const _Float16* __restrict__ Bpack,
                                            float* __restrict__ mv) {
  int tid = blockIdx.x * 256 + threadIdx.x;
  int wid = tid >> 5, lane = tid & 31;
  int wt = wid & 3;
  int h  = (wid >> 2) & 63;
  int dj = (wid >> 8) % 17;
  int b  = wid / 4352;
  int hp = h + dj;
  const _Float16* Ab  = Apack + (((size_t)(b * HH + h) * 4 + wt) * 8) * 512;
  const _Float16* Bb0 = Bpack + (((size_t)(b * HPAD + hp) * 5 + wt) * 8) * 512;
  const _Float16* Bb1 = Bpack + (((size_t)(b * HPAD + hp) * 5 + wt + 1) * 8) * 512;
  v8f acc0 = {}, acc1 = {};
  #pragma unroll
  for (int kc = 0; kc < 8; ++kc) {
    v16h a  = *(const v16h*)(Ab  + kc * 512 + lane * 16);
    v16h b0 = *(const v16h*)(Bb0 + kc * 512 + lane * 16);
    v16h b1 = *(const v16h*)(Bb1 + kc * 512 + lane * 16);
    acc0 = __builtin_amdgcn_wmma_f32_16x16x32_f16(false, a, false, b0, (short)0, acc0, false, false);
    acc1 = __builtin_amdgcn_wmma_f32_16x16x32_f16(false, a, false, b1, (short)0, acc1, false, false);
  }
  int n   = lane & 15;
  int mhi = (lane < 16) ? 0 : 8;
  int w0  = wt * 16;
  #pragma unroll
  for (int r = 0; r < 8; ++r) {
    int M = r + mhi;
    int w = w0 + M;
    int di0 = n - M;                 // column tile 0: wp = w0 + n
    if (di0 >= 0) {                  // di0 <= 15 always
      float v = acc0[r] * (1.f / 256.f);
      v = v > 0.f ? v : 0.1f * v;
      mv[(((size_t)b * NPL + dj * MOO + di0) * HH + h) * WW + w] = v;
    }
    int di1 = n + 16 - M;            // column tile 1: wp = w0 + 16 + n
    if (di1 <= 16) {                 // di1 >= 1 always
      float v = acc1[r] * (1.f / 256.f);
      v = v > 0.f ? v : 0.1f * v;
      mv[(((size_t)b * NPL + dj * MOO + di1) * HH + h) * WW + w] = v;
    }
  }
}

// ---------------------------------------------------------------------------
// k5: zero a u32 region (halo / channel-pad init, deterministic each call)
// ---------------------------------------------------------------------------
__global__ void k_zero(unsigned int* __restrict__ p, int n) {
  int i = blockIdx.x * 256 + threadIdx.x;
  if (i < n) p[i] = 0u;
}

// ---------------------------------------------------------------------------
// k6: 7x7 depthwise attention conv + bias, av = mv*att. LDS-tiled: each
// 256-thread block stages a 22x22 halo tile of one (b,p) plane, then runs
// the 7x7 window entirely from LDS (cuts mv L2 traffic ~24x).
// Output: channels-last halo-padded f16 av_pack[b][66][66][320]
// ---------------------------------------------------------------------------
__global__ __launch_bounds__(256) void k_att(const float* __restrict__ mv,
                                             const float* __restrict__ attw,
                                             const float* __restrict__ attb,
                                             _Float16* __restrict__ avp) {
  __shared__ float tile[22 * 22];
  int bid = blockIdx.x;
  int tx = bid & 3;
  int ty = (bid >> 2) & 3;
  int p  = (bid >> 4) % NPL;
  int b  = bid / (16 * NPL);
  const float* mvb = mv + ((size_t)(b * NPL + p) * HH) * WW;
  const float* wk  = attw + p * 49;

  int h0 = ty * 16 - 3, w0 = tx * 16 - 3;
  for (int i = threadIdx.x; i < 22 * 22; i += 256) {
    int r = i / 22, c = i % 22;
    int hh = h0 + r, ww = w0 + c;
    bool in = ((unsigned)hh < HH) && ((unsigned)ww < WW);
    tile[i] = in ? mvb[hh * WW + ww] : 0.f;
  }
  __syncthreads();

  int lx = threadIdx.x & 15, ly = threadIdx.x >> 4;
  float a = attb[p];
  #pragma unroll
  for (int u = 0; u < 7; ++u)
    #pragma unroll
    for (int v = 0; v < 7; ++v)
      a += tile[(ly + u) * 22 + (lx + v)] * wk[u * 7 + v];

  float av = tile[(ly + 3) * 22 + (lx + 3)] * a;
  int h = ty * 16 + ly, w = tx * 16 + lx;
  avp[(((size_t)b * 66 + h + 1) * 66 + (w + 1)) * 320 + p] = (_Float16)av;
}

// ---------------------------------------------------------------------------
// k7: pack agg1 weights (144,289,3,3) into A-fragments per (octile,uv,kc)
// ---------------------------------------------------------------------------
__global__ void k_packW1(const float* __restrict__ w1, _Float16* __restrict__ W1p) {
  int tid = blockIdx.x * 256 + threadIdx.x;
  if (tid >= 9 * 9 * 10 * 32) return;
  int lane = tid & 31;
  int kc = (tid >> 5) % 10;
  int uv = (tid / 320) % 9;
  int ot = tid / 2880;
  int u = uv / 3, v = uv % 3;
  int oc = ot * 16 + (lane & 15);
  int khi = (lane >= 16) ? 16 : 0;
  _Float16* dst = W1p + ((size_t)((ot * 9 + uv) * 10 + kc)) * 512 + lane * 16;
  #pragma unroll
  for (int j = 0; j < 16; ++j) {
    int ic = kc * 32 + j + khi;
    float val = (ic < NPL) ? w1[(((size_t)oc * NPL + ic) * 3 + u) * 3 + v] : 0.f;
    dst[j] = (_Float16)val;
  }
}

// ---------------------------------------------------------------------------
// k8: GEMM1 (3x3 conv 289->144 as 9 shifted 1x1 GEMMs) + BN + exact GELU.
// Each wave computes TWO 16-column tiles (32 w-positions) sharing one weight
// A-fragment: two independent WMMA acc chains -> pipelined like k_cv.
// -> x1_pack[b][66][66][160] f16 channels-last halo-padded
// ---------------------------------------------------------------------------
__global__ __launch_bounds__(256) void k_gemm1(const _Float16* __restrict__ W1p,
                                               const _Float16* __restrict__ avp,
                                               const float* __restrict__ bn1s,
                                               const float* __restrict__ bn1b,
                                               _Float16* __restrict__ x1p) {
  int tid = blockIdx.x * 256 + threadIdx.x;
  int wid = tid >> 5, lane = tid & 31;
  int ot  = wid % 9;
  int rem = wid / 9;
  int wt2 = rem & 1;          // two 32-column super-tiles: w0 = 0 or 32
  int h   = (rem >> 1) & 63;
  int b   = rem >> 7;
  int w0  = wt2 * 32;
  int n   = lane & 15;
  int kof = (lane < 16) ? 0 : 16;
  v8f acc0 = {}, acc1 = {};
  for (int uv = 0; uv < 9; ++uv) {
    int u = uv / 3, v = uv % 3;
    const _Float16* arow  = W1p + ((size_t)((ot * 9 + uv) * 10)) * 512 + lane * 16;
    const _Float16* brow0 = avp + (((size_t)b * 66 + h + u) * 66 + (w0 + n + v)) * 320 + kof;
    const _Float16* brow1 = brow0 + 16 * 320;   // columns w0+16..w0+31
    #pragma unroll
    for (int kc = 0; kc < 10; ++kc) {
      v16h a   = *(const v16h*)(arow + kc * 512);
      v16h bb0 = *(const v16h*)(brow0 + kc * 32);
      v16h bb1 = *(const v16h*)(brow1 + kc * 32);
      acc0 = __builtin_amdgcn_wmma_f32_16x16x32_f16(false, a, false, bb0, (short)0, acc0, false, false);
      acc1 = __builtin_amdgcn_wmma_f32_16x16x32_f16(false, a, false, bb1, (short)0, acc1, false, false);
    }
  }
  int mhi = (lane < 16) ? 0 : 8;
  const float rinv = rsqrtf(1.f + EPS_BN);
  #pragma unroll
  for (int r = 0; r < 8; ++r) {
    int oc = ot * 16 + r + mhi;
    float sc = bn1s[oc] * rinv, bi = bn1b[oc];
    int w = w0 + n;
    float y = acc0[r] * sc + bi;
    float g = 0.5f * y * (1.f + erff(y * 0.70710678118654752f));
    x1p[(((size_t)b * 66 + h + 1) * 66 + (w + 1)) * 160 + oc] = (_Float16)g;
    w += 16;
    y = acc1[r] * sc + bi;
    g = 0.5f * y * (1.f + erff(y * 0.70710678118654752f));
    x1p[(((size_t)b * 66 + h + 1) * 66 + (w + 1)) * 160 + oc] = (_Float16)g;
  }
}

// ---------------------------------------------------------------------------
// k9: pack agg2 weights (49,144,3,3) into A-fragments (oc padded to 64)
// ---------------------------------------------------------------------------
__global__ void k_packW2(const float* __restrict__ w2, _Float16* __restrict__ W2p) {
  int tid = blockIdx.x * 256 + threadIdx.x;
  if (tid >= 4 * 9 * 5 * 32) return;
  int lane = tid & 31;
  int kc = (tid >> 5) % 5;
  int uv = (tid / 160) % 9;
  int ot = tid / 1440;
  int u = uv / 3, v = uv % 3;
  int oc = ot * 16 + (lane & 15);
  int khi = (lane >= 16) ? 16 : 0;
  _Float16* dst = W2p + ((size_t)((ot * 9 + uv) * 5 + kc)) * 512 + lane * 16;
  #pragma unroll
  for (int j = 0; j < 16; ++j) {
    int ic = kc * 32 + j + khi;
    float val = (oc < C2 && ic < C1) ? w2[(((size_t)oc * C1 + ic) * 3 + u) * 3 + v] : 0.f;
    dst[j] = (_Float16)val;
  }
}

// ---------------------------------------------------------------------------
// k10: GEMM2 (3x3 conv 144->49) + BN + GELU -> d_out NCHW f32.
// Dual 16-column tiles per wave, shared weight A-fragment.
// ---------------------------------------------------------------------------
__global__ __launch_bounds__(256) void k_gemm2(const _Float16* __restrict__ W2p,
                                               const _Float16* __restrict__ x1p,
                                               const float* __restrict__ bn2s,
                                               const float* __restrict__ bn2b,
                                               float* __restrict__ out) {
  int tid = blockIdx.x * 256 + threadIdx.x;
  int wid = tid >> 5, lane = tid & 31;
  int ot  = wid & 3;
  int wt2 = (wid >> 2) & 1;
  int h   = (wid >> 3) & 63;
  int b   = wid >> 9;
  int w0  = wt2 * 32;
  int n   = lane & 15;
  int kof = (lane < 16) ? 0 : 16;
  v8f acc0 = {}, acc1 = {};
  for (int uv = 0; uv < 9; ++uv) {
    int u = uv / 3, v = uv % 3;
    const _Float16* arow  = W2p + ((size_t)((ot * 9 + uv) * 5)) * 512 + lane * 16;
    const _Float16* brow0 = x1p + (((size_t)b * 66 + h + u) * 66 + (w0 + n + v)) * 160 + kof;
    const _Float16* brow1 = brow0 + 16 * 160;
    #pragma unroll
    for (int kc = 0; kc < 5; ++kc) {
      v16h a   = *(const v16h*)(arow + kc * 512);
      v16h bb0 = *(const v16h*)(brow0 + kc * 32);
      v16h bb1 = *(const v16h*)(brow1 + kc * 32);
      acc0 = __builtin_amdgcn_wmma_f32_16x16x32_f16(false, a, false, bb0, (short)0, acc0, false, false);
      acc1 = __builtin_amdgcn_wmma_f32_16x16x32_f16(false, a, false, bb1, (short)0, acc1, false, false);
    }
  }
  int mhi = (lane < 16) ? 0 : 8;
  const float rinv = rsqrtf(1.f + EPS_BN);
  #pragma unroll
  for (int r = 0; r < 8; ++r) {
    int oc = ot * 16 + r + mhi;
    if (oc < C2) {
      float sc = bn2s[oc] * rinv, bi = bn2b[oc];
      int w = w0 + n;
      float y = acc0[r] * sc + bi;
      float g = 0.5f * y * (1.f + erff(y * 0.70710678118654752f));
      out[(((size_t)b * C2 + oc) * HH + h) * WW + w] = g;
      w += 16;
      y = acc1[r] * sc + bi;
      g = 0.5f * y * (1.f + erff(y * 0.70710678118654752f));
      out[(((size_t)b * C2 + oc) * HH + h) * WW + w] = g;
    }
  }
}

// ---------------------------------------------------------------------------
static inline size_t align256(size_t x) { return (x + 255) & ~(size_t)255; }

extern "C" void kernel_launch(void* const* d_in, const int* in_sizes, int n_in,
                              void* d_out, int out_size, void* d_ws, size_t ws_size,
                              hipStream_t stream) {
  const float* f1    = (const float*)d_in[0];
  const float* f2    = (const float*)d_in[1];
  const float* att_w = (const float*)d_in[2];
  const float* att_b = (const float*)d_in[3];
  const float* w1    = (const float*)d_in[4];
  const float* bn1s  = (const float*)d_in[5];
  const float* bn1b  = (const float*)d_in[6];
  const float* w2    = (const float*)d_in[7];
  const float* bn2s  = (const float*)d_in[8];
  const float* bn2b  = (const float*)d_in[9];
  float* out = (float*)d_out;

  char* ws = (char*)d_ws;
  size_t off = 0;
  float*    inv_norm = (float*)(ws + off);    off = align256(off + (size_t)BS * HH * WW * 4);
  _Float16* Apack    = (_Float16*)(ws + off); off = align256(off + (size_t)BS * HH * 4 * 8 * 512 * 2);
  _Float16* Bpack    = (_Float16*)(ws + off); off = align256(off + (size_t)BS * HPAD * 5 * 8 * 512 * 2);
  float*    mv       = (float*)(ws + off);    off = align256(off + (size_t)BS * NPL * HH * WW * 4);
  _Float16* avp      = (_Float16*)(ws + off); off = align256(off + (size_t)BS * 66 * 66 * 320 * 2);
  _Float16* W1p      = (_Float16*)(ws + off); off = align256(off + (size_t)9 * 9 * 10 * 512 * 2);
  _Float16* x1p      = (_Float16*)(ws + off); off = align256(off + (size_t)BS * 66 * 66 * 160 * 2);
  _Float16* W2p      = (_Float16*)(ws + off); off = align256(off + (size_t)4 * 9 * 5 * 512 * 2);

  // stage 0: norms + fragment packing (k_packA reads k_norm's inv_norm; stream order covers it)
  k_norm <<<1024, 256, 0, stream>>>(f1, inv_norm);
  k_packA<<< 512, 256, 0, stream>>>(f1, inv_norm, Apack);
  k_packB<<< 800, 256, 0, stream>>>(f2, Bpack);
  k_packW1<<<102, 256, 0, stream>>>(w1, W1p);
  k_packW2<<< 23, 256, 0, stream>>>(w2, W2p);

  // zero halo / channel padding of channels-last activation buffers
  {
    int n_avp = (int)((size_t)BS * 66 * 66 * 320 * 2 / 4);
    int n_x1p = (int)((size_t)BS * 66 * 66 * 160 * 2 / 4);
    k_zero<<<(n_avp + 255) / 256, 256, 0, stream>>>((unsigned int*)avp, n_avp);
    k_zero<<<(n_x1p + 255) / 256, 256, 0, stream>>>((unsigned int*)x1p, n_x1p);
  }

  // stage 1: cost volume (WMMA) -> mv (post leaky-relu)
  k_cv<<<1088, 256, 0, stream>>>(Apack, Bpack, mv);

  // stage 2: attention depthwise 7x7 (LDS-tiled) + multiply -> av_pack
  k_att<<<BS * NPL * 16, 256, 0, stream>>>(mv, att_w, att_b, avp);

  // stage 3: conv1 (WMMA implicit GEMM, dual-tile) + BN + GELU -> x1_pack
  k_gemm1<<<288, 256, 0, stream>>>(W1p, avp, bn1s, bn1b, x1p);

  // stage 4: conv2 (WMMA implicit GEMM, dual-tile) + BN + GELU -> out
  k_gemm2<<<128, 256, 0, stream>>>(W2p, x1p, bn2s, bn2b, out);
}
